// GraphAttentionLayer_21251498181410
// MI455X (gfx1250) — matrix operand order repaired and checked by reading, not compiled, not run on hardware
//
#include <hip/hip_runtime.h>
#include <math.h>

// Problem constants (reference: B=4, N=4096, D=64)
constexpr int B_ = 4;
constexpr int N_ = 4096;
constexpr int D_ = 64;

typedef __attribute__((ext_vector_type(16))) _Float16 v16h;
typedef __attribute__((ext_vector_type(8)))  _Float16 v8h;
typedef __attribute__((ext_vector_type(8)))  float    v8f;

__device__ __forceinline__ v8f wmma_f16(v16h a, v16h b, v8f c) {
  // D = A(16x32 f16) * B(32x16 f16) + C(16x16 f32)
  return __builtin_amdgcn_wmma_f32_16x16x32_f16(false, a, false, b, (short)0, c,
                                                false, false);
}

// A-fragment (16x32 f16): lane l<16 holds row M=l, K={k0..k0+7, k0+16..k0+23};
// lane l>=16 holds row M=l-16, K={k0+8..k0+15, k0+24..k0+31}. Source: row-major xh.
__device__ __forceinline__ v16h load_a_frag(const _Float16* __restrict__ xh,
                                            int b, int m0, int k0, int lane) {
  const _Float16* row = xh + (size_t)(b * N_ + m0 + (lane & 15)) * D_;
  const _Float16* p = row + k0 + ((lane >> 4) << 3);
  v8h lo = *(const v8h*)p;
  v8h hi = *(const v8h*)(p + 16);
  v16h a;
#pragma unroll
  for (int i = 0; i < 8; ++i) { a[i] = lo[i]; a[i + 8] = hi[i]; }
  return a;
}

// B-fragment (32x16 f16): lane l holds column N=n0+(l&15), K in the same order as
// the A-fragment; strided gather from the transposed copy xhT[b][k][n].
__device__ __forceinline__ v16h load_b_frag(const _Float16* __restrict__ xhT,
                                            int b, int k0, int n0, int lane) {
  const _Float16* col = xhT + (size_t)(b * D_ + k0 + ((lane >> 4) << 3)) * N_ + n0 + (lane & 15);
  v16h v;
#pragma unroll
  for (int i = 0; i < 8; ++i) { v[i] = col[(size_t)i * N_]; v[i + 8] = col[(size_t)(16 + i) * N_]; }
  return v;
}

// ---- prep: f16 copy, f16 transpose, row L2 norms ------------------------------
__global__ void gat_prep(const float* __restrict__ x, _Float16* __restrict__ xh,
                         _Float16* __restrict__ xhT, float* __restrict__ nrm) {
  int tid = blockIdx.x * blockDim.x + threadIdx.x;  // = b*N + n
  if (tid >= B_ * N_) return;
  int b = tid / N_, n = tid % N_;
  const float* row = x + (size_t)tid * D_;
  float ss = 0.f;
  _Float16 h[D_];
#pragma unroll
  for (int d = 0; d < D_; ++d) { float v = row[d]; ss += v * v; h[d] = (_Float16)v; }
  nrm[tid] = sqrtf(ss);
  _Float16* orow = xh + (size_t)tid * D_;
#pragma unroll
  for (int i = 0; i < D_ / 8; ++i) *(v8h*)(orow + i * 8) = *(const v8h*)(h + i * 8);
#pragma unroll
  for (int d = 0; d < D_; ++d) xhT[(size_t)(b * D_ + d) * N_ + n] = h[d];  // coalesced across lanes per d
}

// ---- pass 1: column sums of masked exp(S), softmax axis = n -------------------
// Block: 256 thr = 8 waves, ALL on one 16-wide m strip, 8-way n split.
// Grid: (N/16, B) -> 8192 waves. Software-pipelined: iteration i+1 loads are
// issued before iteration i's exp/accumulate work.
__global__ void __launch_bounds__(256)
gat_colsum(const _Float16* __restrict__ xh, const _Float16* __restrict__ xhT,
           const float* __restrict__ nrm, const int* __restrict__ adj,
           const float* __restrict__ betap, const float* __restrict__ biasp,
           float* __restrict__ colsum) {
  const int b = blockIdx.y;
  const int wave = threadIdx.x >> 5;   // n-eighth
  const int lane = threadIdx.x & 31;
  const int m0 = blockIdx.x * 16;
  const int gb = (lane >> 4) << 3;     // 0 or 8: which m-rows this half-wave owns
  const float beta = betap[0], bias = biasp[0];
  const float LOG2E = 1.4426950408889634f;
  constexpr int ITERS = (N_ / 8) / 16;   // 32

  // A = x rows [m0..m0+15] (fixed for the whole n loop)
  v16h a0 = load_a_frag(xh, b, m0, 0, lane);
  v16h a1 = load_a_frag(xh, b, m0, 32, lane);
  float nm[8];
#pragma unroll
  for (int r = 0; r < 8; ++r) nm[r] = nrm[b * N_ + m0 + gb + r];

  float acc[8];
#pragma unroll
  for (int r = 0; r < 8; ++r) acc[r] = 0.f;

  const int nbase = wave * (N_ / 8);
  const int nl = lane & 15;

  // pipeline prologue: loads for iteration 0
  int n0 = nbase;
  v16h b0 = load_b_frag(xhT, b, 0, n0, lane);
  v16h b1 = load_b_frag(xhT, b, 32, n0, lane);
  const int* ar = adj + ((size_t)b * N_ + n0 + nl) * (size_t)N_ + m0 + gb;
  int4 q0 = *(const int4*)ar;
  int4 q1 = *(const int4*)(ar + 4);
  float nn = nrm[b * N_ + n0 + nl];

  for (int it = 0; it < ITERS; ++it) {
    // rotate current
    v16h cb0 = b0, cb1 = b1;
    int4 cq0 = q0, cq1 = q1;
    float cnn = nn;
    // issue next-iteration loads (clamped; last iter harmlessly reloads it 0)
    const int nit = (it + 1 < ITERS) ? it + 1 : 0;
    const int n1 = nbase + nit * 16;
    b0 = load_b_frag(xhT, b, 0, n1, lane);
    b1 = load_b_frag(xhT, b, 32, n1, lane);
    const int* ar2 = adj + ((size_t)b * N_ + n1 + nl) * (size_t)N_ + m0 + gb;
    q0 = *(const int4*)ar2;
    q1 = *(const int4*)(ar2 + 4);
    nn = nrm[b * N_ + n1 + nl];
    // stream-prefetch adj ~8 iterations (1 KB/row) ahead -> global_prefetch_b8
    const int pf = (it + 8 < ITERS) ? it + 8 : ITERS - 1;
    __builtin_prefetch(adj + ((size_t)b * N_ + nbase + pf * 16 + nl) * (size_t)N_ + m0 + gb, 0, 1);

    // compute with current data
    v8f c = {};
    c = wmma_f16(a0, cb0, c);   // S'[m,n] tile: rows m, cols n
    c = wmma_f16(a1, cb1, c);
    int av[8] = {cq0.x, cq0.y, cq0.z, cq0.w, cq1.x, cq1.y, cq1.z, cq1.w};
#pragma unroll
    for (int r = 0; r < 8; ++r) {
      float s = beta * c[r] * __builtin_amdgcn_rcpf(cnn * nm[r] + bias);
      float e = av[r] ? __builtin_amdgcn_exp2f(s * LOG2E) : 0.f;  // exp(-1e9) == 0
      acc[r] += e;
    }
  }
  // sum over the 16 n-lanes of each half-wave
#pragma unroll
  for (int r = 0; r < 8; ++r) {
    float v = acc[r];
    v += __shfl_xor(v, 1, 32);
    v += __shfl_xor(v, 2, 32);
    v += __shfl_xor(v, 4, 32);
    v += __shfl_xor(v, 8, 32);
    acc[r] = v;
  }
  __shared__ float lds[8][16];
  if ((lane & 15) == 0) {
#pragma unroll
    for (int r = 0; r < 8; ++r) lds[wave][gb + r] = acc[r];
  }
  __syncthreads();
  if (threadIdx.x < 16) {  // deterministic combine of the 8 n-eighths
    int ml = threadIdx.x;
    float s = 0.f;
#pragma unroll
    for (int q = 0; q < 8; ++q) s += lds[q][ml];
    colsum[b * N_ + m0 + ml] = s;
  }
}

__global__ void gat_inv(float* __restrict__ colsum) {
  int i = blockIdx.x * blockDim.x + threadIdx.x;
  if (i < B_ * N_) colsum[i] = 1.0f / colsum[i];
}

// ---- pass 2: out[n,:] = sum_m adj*exp(S[n,m])*rcolsum[m] * x[m,:] -------------
// Block: 256 thr = 8 waves = 2 n-tiles(16) x 4 m-quarters. Grid: (N/32, B)
// -> 4096 waves. S computed TRANSPOSED (rows=m, cols=n): its f32 D-fragment maps
// lane-exactly onto the f16 A-fragment layout of the P*x WMMA. Software-pipelined.
__global__ void __launch_bounds__(256)
gat_out(const _Float16* __restrict__ xh, const _Float16* __restrict__ xhT,
        const float* __restrict__ nrm, const int* __restrict__ adj,
        const float* __restrict__ betap, const float* __restrict__ biasp,
        const float* __restrict__ rcolsum, float* __restrict__ out) {
  const int b = blockIdx.y;
  const int wave = threadIdx.x >> 5;
  const int lane = threadIdx.x & 31;
  const int ntile = wave >> 2;     // 0..1
  const int quarter = wave & 3;    // m quarter
  const int n0 = blockIdx.x * 32 + ntile * 16;
  const int gb = (lane >> 4) << 3;
  const int nl = lane & 15;
  const float beta = betap[0], bias = biasp[0];
  const float LOG2E = 1.4426950408889634f;
  constexpr int ITERS = (N_ / 4) / 32;   // 32

  // B = xT columns n0..n0+15 (fixed for the whole m loop)
  v16h bn0 = load_b_frag(xhT, b, 0, n0, lane);
  v16h bn1 = load_b_frag(xhT, b, 32, n0, lane);
  const float nn = nrm[b * N_ + n0 + nl];

  v8f o[4];
#pragma unroll
  for (int i = 0; i < 4; ++i) { v8f z = {}; o[i] = z; }

  const int mbase = quarter * (N_ / 4);
  const size_t arowbase = ((size_t)b * N_ + n0 + nl) * (size_t)N_;

  // pipeline prologue: loads for iteration 0
  int m0 = mbase;
  v16h am00 = load_a_frag(xh, b, m0, 0, lane);
  v16h am01 = load_a_frag(xh, b, m0, 32, lane);
  v16h am10 = load_a_frag(xh, b, m0 + 16, 0, lane);
  v16h am11 = load_a_frag(xh, b, m0 + 16, 32, lane);
  const int* ar = adj + arowbase + m0 + gb;
  int4 q0 = *(const int4*)ar;
  int4 q1 = *(const int4*)(ar + 4);
  int4 q2 = *(const int4*)(ar + 16);
  int4 q3 = *(const int4*)(ar + 20);

  for (int it = 0; it < ITERS; ++it) {
    const int cm0 = mbase + it * 32;
    // rotate current
    v16h ca00 = am00, ca01 = am01, ca10 = am10, ca11 = am11;
    int4 cq0 = q0, cq1 = q1, cq2 = q2, cq3 = q3;
    // issue next-iteration loads (clamped; last iter harmlessly reloads it 0)
    const int nit = (it + 1 < ITERS) ? it + 1 : 0;
    const int m1 = mbase + nit * 32;
    am00 = load_a_frag(xh, b, m1, 0, lane);
    am01 = load_a_frag(xh, b, m1, 32, lane);
    am10 = load_a_frag(xh, b, m1 + 16, 0, lane);
    am11 = load_a_frag(xh, b, m1 + 16, 32, lane);
    const int* ar2 = adj + arowbase + m1 + gb;
    q0 = *(const int4*)ar2;
    q1 = *(const int4*)(ar2 + 4);
    q2 = *(const int4*)(ar2 + 16);
    q3 = *(const int4*)(ar2 + 20);
    // stream-prefetch adj ~8 iterations (1 KB/row) ahead
    const int pf = (it + 8 < ITERS) ? it + 8 : ITERS - 1;
    __builtin_prefetch(adj + arowbase + mbase + pf * 32 + gb, 0, 1);

    // S' tiles for current m block
    v8f c0 = {}, c1 = {};
    c0 = wmma_f16(ca00, bn0, c0);  c0 = wmma_f16(ca01, bn1, c0);  // S'[cm0..+15 , n]
    c1 = wmma_f16(ca10, bn0, c1);  c1 = wmma_f16(ca11, bn1, c1);  // S'[cm0+16..31, n]

    int av0[8] = {cq0.x, cq0.y, cq0.z, cq0.w, cq1.x, cq1.y, cq1.z, cq1.w};
    int av1[8] = {cq2.x, cq2.y, cq2.z, cq2.w, cq3.x, cq3.y, cq3.z, cq3.w};

    v16h p;  // A-fragment of P(16n x 32m): halves 0..7 <- tile0, 8..15 <- tile1
#pragma unroll
    for (int r = 0; r < 8; ++r) {
      int ma = cm0 + gb + r;
      float s0 = beta * c0[r] * __builtin_amdgcn_rcpf(nn * nrm[b * N_ + ma] + bias);
      float e0 = av0[r] ? __builtin_amdgcn_exp2f(s0 * LOG2E) * rcolsum[b * N_ + ma] : 0.f;
      p[r] = (_Float16)e0;
      int mb = cm0 + 16 + gb + r;
      float s1 = beta * c1[r] * __builtin_amdgcn_rcpf(nn * nrm[b * N_ + mb] + bias);
      float e1 = av1[r] ? __builtin_amdgcn_exp2f(s1 * LOG2E) * rcolsum[b * N_ + mb] : 0.f;
      p[r + 8] = (_Float16)e1;
    }
#pragma unroll
    for (int dseg = 0; dseg < 4; ++dseg) {
      // B[k,d'] = x[cm0+k, dseg*16+d']: lane l holds column d'=l&15, k in A-fragment order (from xhT)
      const _Float16* xp = xhT + (size_t)(b * D_ + dseg * 16 + nl) * N_ + cm0 + gb;
      v8h xlo = *(const v8h*)xp;
      v8h xhi = *(const v8h*)(xp + 16);
      v16h xb;
#pragma unroll
      for (int i = 0; i < 8; ++i) { xb[i] = xlo[i]; xb[i + 8] = xhi[i]; }
      o[dseg] = wmma_f16(p, xb, o[dseg]);
    }
  }

  // deterministic combine of the four m-quarters through LDS, then store
  __shared__ float lds[2 * 3 * 1024];   // 24 KB: partials of quarters 1..3, per n-tile
  if (quarter != 0) {
    float* dst = lds + (ntile * 3 + (quarter - 1)) * 1024;
#pragma unroll
    for (int dseg = 0; dseg < 4; ++dseg)
#pragma unroll
      for (int r = 0; r < 8; ++r)
        dst[(gb + r) * 64 + dseg * 16 + nl] = o[dseg][r];
  }
  __syncthreads();
  if (quarter == 0) {
    const float* p0 = lds + (ntile * 3 + 0) * 1024;
    const float* p1 = lds + (ntile * 3 + 1) * 1024;
    const float* p2 = lds + (ntile * 3 + 2) * 1024;
#pragma unroll
    for (int dseg = 0; dseg < 4; ++dseg)
#pragma unroll
      for (int r = 0; r < 8; ++r) {
        int idx = (gb + r) * 64 + dseg * 16 + nl;
        float v = o[dseg][r] + p0[idx] + p1[idx] + p2[idx];
        out[((size_t)b * N_ + n0 + gb + r) * D_ + dseg * 16 + nl] = v;
      }
  }
}

extern "C" void kernel_launch(void* const* d_in, const int* in_sizes, int n_in,
                              void* d_out, int out_size, void* d_ws, size_t ws_size,
                              hipStream_t stream) {
  const float* x    = (const float*)d_in[0];
  const int*   adj  = (const int*)  d_in[1];
  const float* beta = (const float*)d_in[2];
  const float* bias = (const float*)d_in[3];
  float* out = (float*)d_out;

  // workspace layout (≈4.13 MB): xh f16 | xhT f16 | norms f32 | colsum f32
  char* ws = (char*)d_ws;
  _Float16* xh  = (_Float16*)(ws);
  _Float16* xhT = (_Float16*)(ws + (size_t)B_ * N_ * D_ * 2);
  float* nrm    = (float*)   (ws + (size_t)B_ * N_ * D_ * 4);
  float* colsum = (float*)   (ws + (size_t)B_ * N_ * D_ * 4 + (size_t)B_ * N_ * 4);

  gat_prep<<<(B_ * N_) / 256, 256, 0, stream>>>(x, xh, xhT, nrm);
  gat_colsum<<<dim3(N_ / 16, B_), 256, 0, stream>>>(xh, xhT, nrm, adj, beta, bias, colsum);
  gat_inv<<<(B_ * N_) / 256, 256, 0, stream>>>(colsum);
  gat_out<<<dim3(N_ / 32, B_), 256, 0, stream>>>(xh, xhT, nrm, adj, beta, bias, colsum, out);
}
